// HGT_55405078118539
// MI455X (gfx1250) — matrix operand order repaired and checked by reading, not compiled, not run on hardware
//
#include <hip/hip_runtime.h>
#include <cstdint>

// ---------------------------------------------------------------------------
// Types for CDNA5 (gfx1250) WMMA
// ---------------------------------------------------------------------------
typedef __bf16 bf16;
typedef bf16  v16bf __attribute__((ext_vector_type(16)));
typedef bf16  v8bf  __attribute__((ext_vector_type(8)));
typedef float v8f   __attribute__((ext_vector_type(8)));

union BF16x16 { v16bf v; v8bf h[2]; };

__device__ __forceinline__ v8f wmma_bf16(v16bf a, v16bf b, v8f c) {
#if defined(__HIP_DEVICE_COMPILE__)
  // D = A(16x32 bf16) * B(32x16 bf16) + C(16x16 f32)
  return __builtin_amdgcn_wmma_f32_16x16x32_bf16(false, a, false, b, (short)0, c,
                                                 false, false);
#else
  (void)a; (void)b; return c;
#endif
}

// Async global -> LDS copy of 16 bytes per lane (CDNA5 ASYNCcnt path), with a
// portable load+ds_store fallback if the builtin is not declared.
// Builtin signature (from compiler diagnostic): (int4 AS1*, int4 AS3*, imm, imm)
__device__ __forceinline__ void cp16_g2l(const bf16* __restrict__ g, bf16* __restrict__ l) {
#if defined(__HIP_DEVICE_COMPILE__) && __has_builtin(__builtin_amdgcn_global_load_async_to_lds_b128)
  typedef int v4i __attribute__((ext_vector_type(4)));
  typedef __attribute__((address_space(1))) v4i* gptr;
  typedef __attribute__((address_space(3))) v4i* lptr;
  __builtin_amdgcn_global_load_async_to_lds_b128((gptr)(v4i*)g, (lptr)(v4i*)l, 0, 0);
#else
  *(v8bf*)l = *(const v8bf*)g;
#endif
}

__device__ __forceinline__ void wait_async_lds() {
#if defined(__HIP_DEVICE_COMPILE__)
#if __has_builtin(__builtin_amdgcn_s_wait_asynccnt)
  __builtin_amdgcn_s_wait_asynccnt(0);
#else
  asm volatile("s_wait_asynccnt 0x0" ::: "memory");
#endif
#endif
}

// Monotone float <-> signed-int mapping for atomicMax-based segment max
__device__ __forceinline__ int   f2ord(float f){ int i=__float_as_int(f); return i>=0 ? i : (i ^ 0x7fffffff); }
__device__ __forceinline__ float ord2f(int k){ int i = (k>=0) ? k : (k ^ 0x7fffffff); return __int_as_float(i); }

// ---------------------------------------------------------------------------
// Elementwise / conversion kernels
// ---------------------------------------------------------------------------
__global__ void k_f2b(const float* __restrict__ in, bf16* __restrict__ out, long n) {
  long i = (long)blockIdx.x * blockDim.x + threadIdx.x;
  if (i < n) out[i] = (bf16)in[i];
}

// W[K][N] fp32 -> Wt[N][K] bf16 (so a WMMA B-column is contiguous)
__global__ void k_transW(const float* __restrict__ W, bf16* __restrict__ Wt, int K, int N) {
  long i = (long)blockIdx.x * blockDim.x + threadIdx.x;
  if (i >= (long)K * N) return;
  int k = (int)(i / N), n = (int)(i % N);
  Wt[(size_t)n * K + k] = (bf16)W[i];
}

// rel[8][32][32] (d,e) fp32 -> relT[8][32][32] (e,d) bf16
__global__ void k_relTk(const float* __restrict__ R, bf16* __restrict__ T) {
  int i = blockIdx.x * blockDim.x + threadIdx.x;
  if (i >= 8192) return;
  int h = i >> 10, r = i & 1023, d = r >> 5, e = r & 31;
  T[h * 1024 + e * 32 + d] = (bf16)R[i];
}

__global__ void k_fillf(float* __restrict__ p, float v, long n) {
  long i = (long)blockIdx.x * blockDim.x + threadIdx.x;
  if (i < n) p[i] = v;
}
__global__ void k_filli(int* __restrict__ p, int v, long n) {
  long i = (long)blockIdx.x * blockDim.x + threadIdx.x;
  if (i < n) p[i] = v;
}

__global__ void k_gelu(const float* __restrict__ in, bf16* __restrict__ out, long n) {
  long i = (long)blockIdx.x * blockDim.x + threadIdx.x;
  if (i >= n) return;
  float x = in[i];
  float g = 0.5f * x * (1.0f + erff(x * 0.70710678118654752f)); // exact GELU
  out[i] = (bf16)g;
}

__global__ void k_skip(const float* __restrict__ o, const float* __restrict__ xprev,
                       const float* __restrict__ skip, float* __restrict__ outF,
                       bf16* __restrict__ outB, long n) {
  long i = (long)blockIdx.x * blockDim.x + threadIdx.x;
  if (i >= n) return;
  float s = 1.0f / (1.0f + __expf(-skip[0]));
  float v = s * o[i] + (1.0f - s) * xprev[i];
  outF[i] = v;
  if (outB) outB[i] = (bf16)v;
}

// ---------------------------------------------------------------------------
// WMMA GEMM: out[M,N] = act(A[M,K]bf16 @ Bt[N,K]^T bf16 + bias)
// block = 256 threads (8 waves); each wave: 16 rows x 64 cols (4 wmma accs).
// B tile is staged in LDS (double-buffered, async global->LDS); A fragments
// are software-pipelined so WMMAs never wait on the current global loads.
// grid = (ceil(M/128), N/64). LDS row stride padded to 40 bf16 (20 dwords)
// to spread ds_load_b128 across banks.
// ---------------------------------------------------------------------------
#define BSTRIDE 40

__global__ __launch_bounds__(256) void k_gemm(
    const bf16* __restrict__ A, const bf16* __restrict__ Bt,
    const float* __restrict__ bias, float* __restrict__ outF, bf16* __restrict__ outB,
    int M, int N, int K, int relu)
{
  __shared__ bf16 Bs[2 * 64 * BSTRIDE];                // 10 KB, double-buffered

  const int wave = threadIdx.x >> 5;
  const int lane = threadIdx.x & 31;
  const int l16  = lane & 15;
  const int half = lane >> 4;
  const int row0 = blockIdx.x * 128 + wave * 16;
  const int col0 = blockIdx.y * 64;

  // staging map: 256 threads cover 64 rows x 4 x 16B chunks (one 64x32 tile)
  const int sn = threadIdx.x >> 2;                     // row 0..63
  const int sc = (threadIdx.x & 3) * 8;                // chunk offset in bf16
  const bf16* Bg = Bt + (size_t)(col0 + sn) * K + sc;  // + k0 per tile
  bf16* Bl = Bs + sn * BSTRIDE + sc;                   // + buf select per tile

  int ar = row0 + l16; ar = (ar < M) ? ar : (M - 1);   // clamp, no divergence
  const bf16* Ap = A + (size_t)ar * K;

  // prologue: stage B tile 0, preload A fragment 0
  cp16_g2l(Bg, Bl);
  BF16x16 a_cur, a_nxt;
  a_cur.h[0] = *(const v8bf*)(Ap + half * 16);
  a_cur.h[1] = *(const v8bf*)(Ap + half * 16 + 8);
  a_nxt = a_cur;
  wait_async_lds();
  __syncthreads();

  v8f acc[4] = {};
  const int niter = K >> 5;
  for (int it = 0; it < niter; ++it) {
    if (it + 1 < niter) {                              // issue next tile early
      cp16_g2l(Bg + ((it + 1) << 5), Bl + ((it + 1) & 1) * 64 * BSTRIDE);
      const int kn = (it + 1) << 5;
      a_nxt.h[0] = *(const v8bf*)(Ap + kn + half * 16);
      a_nxt.h[1] = *(const v8bf*)(Ap + kn + half * 16 + 8);
    }
    const bf16* Bb = Bs + (it & 1) * 64 * BSTRIDE;
#pragma unroll
    for (int j = 0; j < 4; ++j) {
      const bf16* bp = Bb + (j * 16 + l16) * BSTRIDE + half * 16;
      BF16x16 b;
      b.h[0] = *(const v8bf*)(bp);
      b.h[1] = *(const v8bf*)(bp + 8);
      acc[j] = wmma_bf16(a_cur.v, b.v, acc[j]);
    }
    a_cur = a_nxt;
    if (it + 1 < niter) { wait_async_lds(); __syncthreads(); }
  }

#pragma unroll
  for (int j = 0; j < 4; ++j) {
    const int n = col0 + j * 16 + l16;
    const float bv = bias ? bias[n] : 0.0f;
#pragma unroll
    for (int r = 0; r < 8; ++r) {
      const int m = row0 + r + half * 8;               // C/D: vgpr r -> row r (+8 hi half)
      if (m < M) {
        float v = acc[j][r] + bv;
        if (relu) v = fmaxf(v, 0.0f);
        if (outF) outF[(size_t)m * N + n] = v;
        if (outB) outB[(size_t)m * N + n] = (bf16)v;
      }
    }
  }
}

// ---------------------------------------------------------------------------
// Per-head relation transform: out[n,h,:] = X[n,h,:] @ T_h   (T_h is 32x32)
// block = 256 (8 waves, wave w == head w), 16 nodes per block, 2 WMMA per wave
// ---------------------------------------------------------------------------
__global__ __launch_bounds__(256) void k_rel(
    const bf16* __restrict__ X, const bf16* __restrict__ T,
    float* __restrict__ out, int Nn)
{
  const int head = threadIdx.x >> 5;
  const int lane = threadIdx.x & 31;
  const int l16 = lane & 15, half = lane >> 4;
  const int n0 = blockIdx.x * 16;
  int nr = n0 + l16; nr = (nr < Nn) ? nr : (Nn - 1);

  const bf16* Xp = X + (size_t)nr * 256 + head * 32;
  BF16x16 a;
  a.h[0] = *(const v8bf*)(Xp + half * 16);
  a.h[1] = *(const v8bf*)(Xp + half * 16 + 8);

  const bf16* Tb = T + head * 1024;                    // [32 cols][32 k] bf16
  v8f acc[2] = {};
#pragma unroll
  for (int j = 0; j < 2; ++j) {
    BF16x16 b; b.v = *(const v16bf*)(Tb + (j * 16 + l16) * 32 + half * 16);
    acc[j] = wmma_bf16(a.v, b.v, acc[j]);
  }
#pragma unroll
  for (int j = 0; j < 2; ++j)
#pragma unroll
    for (int r = 0; r < 8; ++r) {
      const int m = n0 + r + half * 8;
      if (m < Nn) out[(size_t)m * 256 + head * 32 + j * 16 + l16] = acc[j][r];
    }
}

// ---------------------------------------------------------------------------
// Edge phase: one thread per (edge, head)
// ---------------------------------------------------------------------------
__global__ void k_score(const float* __restrict__ q, const float* __restrict__ krel,
                        const int* __restrict__ src, const int* __restrict__ dst,
                        const float* __restrict__ p, float* __restrict__ score,
                        int* __restrict__ mkey, int E)
{
  long t = (long)blockIdx.x * blockDim.x + threadIdx.x;
  if (t >= (long)E * 8) return;
  int e = (int)(t >> 3), h = (int)(t & 7);
  int s = src[e], d = dst[e];
  const float4* qp = (const float4*)(q + (size_t)d * 256 + h * 32);
  const float4* kp = (const float4*)(krel + (size_t)s * 256 + h * 32);
  float acc = 0.0f;
#pragma unroll
  for (int i = 0; i < 8; ++i) {
    float4 a = qp[i], b = kp[i];
    acc += a.x * b.x + a.y * b.y + a.z * b.z + a.w * b.w;
  }
  acc *= p[h] * 0.17677669529663687f;                  // 1/sqrt(32)
  score[t] = acc;
  atomicMax(&mkey[d * 8 + h], f2ord(acc));
}

__global__ void k_exp(float* __restrict__ score, const int* __restrict__ mkey,
                      float* __restrict__ ssum, const int* __restrict__ dst, int E)
{
  long t = (long)blockIdx.x * blockDim.x + threadIdx.x;
  if (t >= (long)E * 8) return;
  int e = (int)(t >> 3), h = (int)(t & 7);
  int d = dst[e];
  float ex = __expf(score[t] - ord2f(mkey[d * 8 + h]));
  score[t] = ex;
  atomicAdd(&ssum[d * 8 + h], ex);
}

__global__ void k_agg(const float* __restrict__ score, const float* __restrict__ ssum,
                      const float* __restrict__ vrel, const int* __restrict__ src,
                      const int* __restrict__ dst, float* __restrict__ agg, int E)
{
  long t = (long)blockIdx.x * blockDim.x + threadIdx.x;
  if (t >= (long)E * 8) return;
  int e = (int)(t >> 3), h = (int)(t & 7);
  int s = src[e], d = dst[e];
  float alpha = score[t] / (ssum[d * 8 + h] + 1e-16f);
  const float* vp = vrel + (size_t)s * 256 + h * 32;
  float* ap = agg + (size_t)d * 256 + h * 32;
#pragma unroll
  for (int i = 0; i < 32; ++i) atomicAdd(ap + i, alpha * vp[i]);
}

// ---------------------------------------------------------------------------
// Host orchestration
// ---------------------------------------------------------------------------
extern "C" void kernel_launch(void* const* d_in, const int* in_sizes, int n_in,
                              void* d_out, int out_size, void* d_ws, size_t ws_size,
                              hipStream_t stream)
{
  (void)n_in; (void)out_size; (void)ws_size;
  constexpr int F = 512, H = 256;
  const float* x_mi = (const float*)d_in[0];
  const float* x_ge = (const float*)d_in[1];
  const int* e_src[3] = { (const int*)d_in[2], (const int*)d_in[4], (const int*)d_in[6] };
  const int* e_dst[3] = { (const int*)d_in[3], (const int*)d_in[5], (const int*)d_in[7] };
  const int E    = in_sizes[2];
  const int N_MI = in_sizes[0] / F;
  const int N_GE = in_sizes[1] / F;
  const long NT  = (long)N_MI + N_GE;

  // params flattened in JAX sorted-dict order starting at d_in[8]:
  // layers[0..1] (a,k,q,rel{gg,gm,mg},skip,v), then lin. 27 leaves per layer.
  auto P = [&](int i) -> const float* { return (const float*)d_in[8 + i]; };

  // ---- workspace arena ----
  char* base = (char*)d_ws; size_t off = 0;
  auto alloc = [&](size_t bytes) -> void* {
    void* p = base + off; off = (off + bytes + 255) & ~(size_t)255; return p;
  };
  bf16*  xbin   = (bf16*) alloc((size_t)NT * F * 2);
  bf16*  WtLinG = (bf16*) alloc((size_t)H * F * 2);
  bf16*  WtLinM = (bf16*) alloc((size_t)H * F * 2);
  bf16*  WtG[4]; bf16* WtM[4];                        // k,q,v,a (reused per layer)
  for (int i = 0; i < 4; ++i) { WtG[i] = (bf16*)alloc((size_t)H*H*2); WtM[i] = (bf16*)alloc((size_t)H*H*2); }
  bf16*  relTa = (bf16*) alloc(8192 * 2);
  bf16*  relTm = (bf16*) alloc(8192 * 2);
  float* curx  = (float*)alloc((size_t)NT * H * 4);
  bf16*  curxb = (bf16*) alloc((size_t)NT * H * 2);
  float* qbuf  = (float*)alloc((size_t)NT * H * 4);
  bf16*  kbuf  = (bf16*) alloc((size_t)NT * H * 2);
  bf16*  vbuf  = (bf16*) alloc((size_t)NT * H * 2);
  float* krel  = (float*)alloc((size_t)N_GE * H * 4);
  float* vrel  = (float*)alloc((size_t)N_GE * H * 4);
  float* score = (float*)alloc((size_t)E * 8 * 4);
  int*   mkey  = (int*)  alloc((size_t)N_GE * 8 * 4);
  float* ssum  = (float*)alloc((size_t)N_GE * 8 * 4);
  float* agg   = (float*)alloc((size_t)NT * H * 4);
  bf16*  gelub = (bf16*) alloc((size_t)NT * H * 2);
  float* obuf  = (float*)alloc((size_t)NT * H * 4);

  auto cdiv = [](long a, long b) -> unsigned { return (unsigned)((a + b - 1) / b); };
  auto gemm = [&](const bf16* A, const bf16* Bt, const float* bias,
                  float* oF, bf16* oB, int M, int N, int K, int relu) {
    dim3 g(cdiv(M, 128), N / 64);
    k_gemm<<<g, 256, 0, stream>>>(A, Bt, bias, oF, oB, M, N, K, relu);
  };

  const long nmiF = (long)N_MI * F, ngeF = (long)N_GE * F;
  const long nmiH = (long)N_MI * H, ngeH = (long)N_GE * H;

  // ---- input conversion + input projection (fused bias+relu) ----
  k_f2b<<<cdiv(nmiF, 256), 256, 0, stream>>>(x_mi, xbin, nmiF);
  k_f2b<<<cdiv(ngeF, 256), 256, 0, stream>>>(x_ge, xbin + nmiF, ngeF);
  k_transW<<<cdiv((long)F * H, 256), 256, 0, stream>>>(P(54), WtLinG, F, H); // lin gene W
  k_transW<<<cdiv((long)F * H, 256), 256, 0, stream>>>(P(56), WtLinM, F, H); // lin miRNA W
  gemm(xbin,        WtLinM, P(57), curx,        curxb,        N_MI, H, F, 1);
  gemm(xbin + nmiF, WtLinG, P(55), curx + nmiH, curxb + nmiH, N_GE, H, F, 1);

  const int relOf[3][3]  = { {18,19,20}, {15,16,17}, {12,13,14} }; // mg, gm, gg
  const int srcIsGene[3] = { 0, 1, 1 };
  const int dstIsGene[3] = { 1, 0, 1 };

  for (int l = 0; l < 2; ++l) {
    const int bs = 27 * l;
    // weight transposes (k,q,v,a) x (gene, miRNA)
    k_transW<<<cdiv((long)H*H,256),256,0,stream>>>(P(bs+4),  WtG[0], H, H); // k gene
    k_transW<<<cdiv((long)H*H,256),256,0,stream>>>(P(bs+6),  WtM[0], H, H); // k mi
    k_transW<<<cdiv((long)H*H,256),256,0,stream>>>(P(bs+8),  WtG[1], H, H); // q gene
    k_transW<<<cdiv((long)H*H,256),256,0,stream>>>(P(bs+10), WtM[1], H, H); // q mi
    k_transW<<<cdiv((long)H*H,256),256,0,stream>>>(P(bs+23), WtG[2], H, H); // v gene
    k_transW<<<cdiv((long)H*H,256),256,0,stream>>>(P(bs+25), WtM[2], H, H); // v mi
    k_transW<<<cdiv((long)H*H,256),256,0,stream>>>(P(bs+0),  WtG[3], H, H); // a gene
    k_transW<<<cdiv((long)H*H,256),256,0,stream>>>(P(bs+2),  WtM[3], H, H); // a mi

    float* q_mi = qbuf;  float* q_ge = qbuf + nmiH;
    bf16*  k_mi = kbuf;  bf16*  k_ge = kbuf + nmiH;
    bf16*  v_mi = vbuf;  bf16*  v_ge = vbuf + nmiH;

    gemm(curxb,        WtM[1], P(bs+11), q_mi,    nullptr, N_MI, H, H, 0);
    gemm(curxb + nmiH, WtG[1], P(bs+9),  q_ge,    nullptr, N_GE, H, H, 0);
    gemm(curxb,        WtM[0], P(bs+7),  nullptr, k_mi,    N_MI, H, H, 0);
    gemm(curxb + nmiH, WtG[0], P(bs+5),  nullptr, k_ge,    N_GE, H, H, 0);
    gemm(curxb,        WtM[2], P(bs+26), nullptr, v_mi,    N_MI, H, H, 0);
    gemm(curxb + nmiH, WtG[2], P(bs+24), nullptr, v_ge,    N_GE, H, H, 0);

    k_fillf<<<cdiv(NT * H, 256), 256, 0, stream>>>(agg, 0.0f, NT * H);

    for (int t = 0; t < 3; ++t) {
      const bf16*  ksrc = srcIsGene[t] ? k_ge : k_mi;
      const bf16*  vsrc = srcIsGene[t] ? v_ge : v_mi;
      const int    Ns   = srcIsGene[t] ? N_GE : N_MI;
      const float* qdst = dstIsGene[t] ? q_ge : q_mi;
      const int    Nd   = dstIsGene[t] ? N_GE : N_MI;
      float*       aggd = dstIsGene[t] ? (agg + nmiH) : agg;

      k_relTk<<<32, 256, 0, stream>>>(P(bs + relOf[t][0]), relTa);
      k_relTk<<<32, 256, 0, stream>>>(P(bs + relOf[t][1]), relTm);
      k_rel<<<cdiv(Ns, 16), 256, 0, stream>>>(ksrc, relTa, krel, Ns);
      k_rel<<<cdiv(Ns, 16), 256, 0, stream>>>(vsrc, relTm, vrel, Ns);

      k_filli<<<cdiv((long)Nd * 8, 256), 256, 0, stream>>>(mkey, (int)0x807FFFFF, (long)Nd * 8);
      k_fillf<<<cdiv((long)Nd * 8, 256), 256, 0, stream>>>(ssum, 0.0f, (long)Nd * 8);

      const unsigned gE = cdiv((long)E * 8, 256);
      k_score<<<gE, 256, 0, stream>>>(qdst, krel, e_src[t], e_dst[t],
                                      P(bs + relOf[t][2]), score, mkey, E);
      k_exp<<<gE, 256, 0, stream>>>(score, mkey, ssum, e_dst[t], E);
      k_agg<<<gE, 256, 0, stream>>>(score, ssum, vrel, e_src[t], e_dst[t], aggd, E);
    }

    // gelu -> bf16, 'a' projection, sigmoid-skip combine
    k_gelu<<<cdiv(NT * H, 256), 256, 0, stream>>>(agg, gelub, NT * H);
    gemm(gelub,        WtM[3], P(bs+3), obuf,        nullptr, N_MI, H, H, 0);
    gemm(gelub + nmiH, WtG[3], P(bs+1), obuf + nmiH, nullptr, N_GE, H, H, 0);

    const bool last = (l == 1);
    float* oF_mi = last ? (float*)d_out          : curx;
    float* oF_ge = last ? (float*)d_out + nmiH   : curx + nmiH;
    bf16*  oB_mi = last ? nullptr : curxb;
    bf16*  oB_ge = last ? nullptr : (curxb + nmiH);
    k_skip<<<cdiv(nmiH, 256), 256, 0, stream>>>(obuf,        curx,        P(bs+22), oF_mi, oB_mi, nmiH);
    k_skip<<<cdiv(ngeH, 256), 256, 0, stream>>>(obuf + nmiH, curx + nmiH, P(bs+21), oF_ge, oB_ge, ngeH);
  }
}
